// Model_31069793419696
// MI455X (gfx1250) — matrix operand order repaired and checked by reading, not compiled, not run on hardware
//
#include <hip/hip_runtime.h>

// ---------------- constants ----------------
static const int NN = 100000;   // nodes
static const int NE = 400000;   // edges
static const int NG = 4096;     // graphs
static const int DD = 300;      // emb dim
static const int LL = 5;        // layers
static const int NP = 100096;   // nodes padded to 128
static const int DP = 320;      // D padded to 32
static const int HP = 640;      // 2*D padded to 64

typedef __attribute__((ext_vector_type(16))) __bf16 v16bf;
typedef __attribute__((ext_vector_type(8)))  float  v8f;

static __device__ __forceinline__ unsigned short f2bf(float f) {
  unsigned u = __float_as_uint(f);
  unsigned r = (u + 0x7FFFu + ((u >> 16) & 1u)) >> 16;   // round-to-nearest-even
  return (unsigned short)r;
}

static inline int cdiv(long a, int b) { return (int)((a + b - 1) / b); }

// ============ WMMA fragment swizzle helpers (CDNA5 wave32 layouts) ============
// A 16x32 bf16 fragment: lanes0-15 hold M=lane, K {0..7,16..23}; lanes16-31 K {8..15,24..31}
// elem->K: lane<16: K = (i<8)? i : i+8 ; lane>=16: K = (i<8)? i+8 : i+16
// B 32x16 bf16 fragment: lanes0-15 hold N=lane, K {0..7,16..23}; lanes16-31 K {8..15,24..31}
// Swizzled storage: tile-major, then 32 lanes x 16 contiguous bf16 (32B per lane).
static __device__ __forceinline__ size_t a_swz(int m, int k, int K) {
  int kk = k & 31;
  int lanep = (m & 15) | (((kk >> 3) & 1) << 4);
  int elem  = (kk & 7) + ((kk & 16) ? 8 : 0);
  return (((size_t)(m >> 4) * (K >> 5) + (k >> 5)) * 32 + lanep) * 16 + elem;
}
static __device__ __forceinline__ size_t b_swz(int k, int n, int N) {
  int kk = k & 31;
  int lane = (n & 15) | (((kk >> 3) & 1) << 4);
  int elem = (kk & 7) + ((kk & 16) ? 8 : 0);
  return (((size_t)(k >> 5) * (N >> 4) + (n >> 4)) * 32 + lane) * 16 + elem;
}

// ---------------- generic kernels ----------------
__global__ void k_zero(float* p, int n) {
  int i = blockIdx.x * blockDim.x + threadIdx.x;
  if (i < n) p[i] = 0.f;
}

// weights f32 [S][R][C] -> bf16 zero-padded, B-fragment swizzled [S][Kp][Np]
__global__ void k_cvt_b(const float* __restrict__ src, unsigned short* __restrict__ dst,
                        int S, int R, int C, int Kp, int Np) {
  long i = (long)blockIdx.x * blockDim.x + threadIdx.x;
  long total = (long)S * Kp * Np;
  if (i >= total) return;
  int n = (int)(i % Np);
  long t = i / Np;
  int k = (int)(t % Kp);
  int s = (int)(t / Kp);
  float v = (k < R && n < C) ? src[((long)s * R + k) * C + n] : 0.f;
  dst[(size_t)s * Kp * Np + b_swz(k, n, Np)] = f2bf(v);
}

// bias f32 [S][C] -> zero-padded f32 [S][Cp]
__global__ void k_padbias(const float* __restrict__ src, float* __restrict__ dst,
                          int S, int C, int Cp) {
  int i = blockIdx.x * blockDim.x + threadIdx.x;
  if (i >= S * Cp) return;
  int c = i % Cp, s = i / Cp;
  dst[i] = (c < C) ? src[s * C + c] : 0.f;
}

// ---------------- graph kernels ----------------
__global__ void k_init_h(const int* __restrict__ x, const float* __restrict__ ae1,
                         const float* __restrict__ ae2, float* __restrict__ h) {
  int i = blockIdx.x * blockDim.x + threadIdx.x;
  if (i >= NN * DD) return;
  int n = i / DD, d = i - n * DD;
  int a = x[2 * n], c = x[2 * n + 1];
  h[(size_t)n * DP + d] = ae1[(size_t)a * DD + d] + ae2[(size_t)c * DD + d];
}

__global__ void k_agg_init(const float* __restrict__ h, const float* __restrict__ sl1,
                           const float* __restrict__ sl2, float* __restrict__ agg) {
  int i = blockIdx.x * blockDim.x + threadIdx.x;
  if (i >= NN * DD) return;
  int n = i / DD, d = i - n * DD;
  agg[(size_t)n * DP + d] = h[(size_t)n * DP + d] + sl1[d] + sl2[d];
}

__global__ void k_edge(const int* __restrict__ ei, const int* __restrict__ ea,
                       const float* __restrict__ h, const float* __restrict__ e1t,
                       const float* __restrict__ e2t, float* __restrict__ agg) {
  int i = blockIdx.x * blockDim.x + threadIdx.x;
  if (i >= NE * 75) return;
  int e = i / 75, c = i - e * 75, d = c * 4;
  int s  = ei[e], t2 = ei[NE + e];
  int a0 = ea[2 * e], a1 = ea[2 * e + 1];
  const float4 hv = *(const float4*)(h   + (size_t)s  * DP + d);
  const float4 v1 = *(const float4*)(e1t + (size_t)a0 * DD + d);
  const float4 v2 = *(const float4*)(e2t + (size_t)a1 * DD + d);
  float* dst = agg + (size_t)t2 * DP + d;
  atomicAdd(dst + 0, hv.x + v1.x + v2.x);
  atomicAdd(dst + 1, hv.y + v1.y + v2.y);
  atomicAdd(dst + 2, hv.z + v1.z + v2.z);
  atomicAdd(dst + 3, hv.w + v1.w + v2.w);
}

// ---------------- batchnorm ----------------
__global__ void k_bn_stats(const float* __restrict__ z, float* __restrict__ sum,
                           float* __restrict__ sq) {
  __shared__ float ss[DD];
  __shared__ float s2[DD];
  for (int j = threadIdx.x; j < DD; j += blockDim.x) { ss[j] = 0.f; s2[j] = 0.f; }
  __syncthreads();
  const long total = (long)NN * DD;
  for (long i = (long)blockIdx.x * blockDim.x + threadIdx.x; i < total;
       i += (long)gridDim.x * blockDim.x) {
    int n = (int)(i / DD), d = (int)(i - (long)n * DD);
    float v = z[(size_t)n * DP + d];
    atomicAdd(&ss[d], v);
    atomicAdd(&s2[d], v * v);
  }
  __syncthreads();
  for (int j = threadIdx.x; j < DD; j += blockDim.x) {
    atomicAdd(&sum[j], ss[j]);
    atomicAdd(&sq[j],  s2[j]);
  }
}

__global__ void k_bn_apply(const float* __restrict__ z, const float* __restrict__ sum,
                           const float* __restrict__ sq, const float* __restrict__ g,
                           const float* __restrict__ b, float* __restrict__ h, int relu) {
  int i = blockIdx.x * blockDim.x + threadIdx.x;
  if (i >= NN * DD) return;
  int n = i / DD, d = i - n * DD;
  const float invN = 1.f / (float)NN;
  float mu  = sum[d] * invN;
  float var = sq[d] * invN - mu * mu;
  float v = (z[(size_t)n * DP + d] - mu) * rsqrtf(var + 1e-5f) * g[d] + b[d];
  if (relu) v = fmaxf(v, 0.f);
  h[(size_t)n * DP + d] = v;
}

// ---------------- pooling / branch ----------------
__global__ void k_counts(const int* __restrict__ bid, float* __restrict__ counts) {
  int n = blockIdx.x * blockDim.x + threadIdx.x;
  if (n < NN) atomicAdd(&counts[bid[n]], 1.f);
}

__global__ void k_pool(const int* __restrict__ bid, const float* __restrict__ h,
                       float* __restrict__ pooled) {
  int i = blockIdx.x * blockDim.x + threadIdx.x;
  if (i >= NN * 75) return;
  int n = i / 75, c = i - n * 75, d = c * 4;
  int g = bid[n];
  const float4 hv = *(const float4*)(h + (size_t)n * DP + d);
  float* dst = pooled + (size_t)g * DP + d;
  atomicAdd(dst + 0, hv.x);
  atomicAdd(dst + 1, hv.y);
  atomicAdd(dst + 2, hv.z);
  atomicAdd(dst + 3, hv.w);
}

__global__ void k_pooldiv(const float* __restrict__ counts, float* __restrict__ pooled) {
  int i = blockIdx.x * blockDim.x + threadIdx.x;
  if (i >= NG * DD) return;
  int g = i / DD, d = i - g * DD;
  pooled[(size_t)g * DP + d] = pooled[(size_t)g * DP + d] / fmaxf(counts[g], 1.f);
}

__global__ void k_gather(const int* __restrict__ dang, const float* __restrict__ h,
                         float* __restrict__ hd) {
  int i = blockIdx.x * blockDim.x + threadIdx.x;
  if (i >= NG * DP) return;
  int g = i / DP, d = i - g * DP;
  hd[i] = h[(size_t)dang[g] * DP + d];
}

// row-wise L2 normalize of (o1+o2);
// mode 0: write bf16 A-fragment-swizzled (f0, logits A operand, K=DP)
// mode 1: write bf16 B-fragment-swizzled (f1^T, logits B operand, N=NG)
__global__ void k_norm(const float* __restrict__ o1, const float* __restrict__ o2,
                       unsigned short* __restrict__ f0b, unsigned short* __restrict__ f1t,
                       int mode) {
  __shared__ float red[256];
  int g = blockIdx.x, t = threadIdx.x;
  float p = 0.f;
  for (int d = t; d < DD; d += 256) {
    float s = o1[(size_t)g * DP + d] + o2[(size_t)g * DP + d];
    p += s * s;
  }
  red[t] = p;
  __syncthreads();
  for (int o = 128; o > 0; o >>= 1) {
    if (t < o) red[t] += red[t + o];
    __syncthreads();
  }
  float rn = rsqrtf(red[0]);
  for (int d = t; d < DP; d += 256) {
    float v = (d < DD) ? (o1[(size_t)g * DP + d] + o2[(size_t)g * DP + d]) * rn : 0.f;
    unsigned short bv = f2bf(v);
    if (mode == 0) f0b[a_swz(g, d, DP)] = bv;
    else           f1t[b_swz(d, g, NG)] = bv;
  }
}

// ---------------- WMMA GEMM ----------------
// C[M,N] = act((A[M,K] @ B[K,N] + bias) * scale)
// AMODE 0: A is f32 row-major (lda), staged through LDS with f32->bf16 convert.
// AMODE 1: A is bf16 in A-fragment swizzle -> loaded straight into WMMA regs (no LDS).
// B is always bf16 in B-fragment swizzle -> loaded straight into WMMA regs.
// OUTM 1: f32 row-major (ldc).  OUTM 2: bf16 A-fragment swizzle (for next GEMM's A).
// Tile: 128x64 per block (8 waves; each wave 32x32 = 2x2 WMMA 16x16x32 frags).
template <int AMODE, bool RELU, bool HASBIAS, int OUTM>
__global__ __launch_bounds__(256) void k_gemm(
    const void* __restrict__ Av, const unsigned short* __restrict__ Bsw,
    const float* __restrict__ bias, float* __restrict__ Cf,
    unsigned short* __restrict__ Cb,
    int M, int N, int K, int lda, int ldc, float scale) {
  extern __shared__ __attribute__((aligned(16))) unsigned short As[];  // [128][40] when AMODE==0

  const int tid  = threadIdx.x;
  const int lane = tid & 31;
  const int wid  = tid >> 5;
  const int wm   = wid & 3;   // M quadrant (32 rows)
  const int wn   = wid >> 2;  // N half (32 cols)
  const int m0   = blockIdx.y * 128;
  const int n0   = blockIdx.x * 64;

  const float*          Af  = (const float*)Av;
  const unsigned short* Asw = (const unsigned short*)Av;
  const int KT  = K >> 5;   // A swizzle k-tiles per m-tile row
  const int NT4 = N >> 4;   // B swizzle n-tiles per k-tile row

  v8f acc[2][2] = {};

  for (int k0 = 0; k0 < K; k0 += 32) {
    union Frag { v16bf v; uint4 q[2]; } fa[2], fb[2];
    const int kt = k0 >> 5;

    if constexpr (AMODE == 0) {
      // stage A tile 128x32: f32 global -> bf16 LDS (stride 40 halves, 16B-aligned rows)
#pragma unroll
      for (int it = 0; it < 4; ++it) {
        int q  = it * 256 + tid;  // 1024 float4 slots
        int r  = q >> 3;
        int c4 = q & 7;
        const float* sp = Af + (size_t)(m0 + r) * lda + (k0 + c4 * 4);
        const float4 v = *(const float4*)sp;
        __builtin_prefetch(sp + 32, 0, 1);   // next K tile -> global_prefetch_b8
        unsigned short* dst = &As[r * 40 + c4 * 4];
        dst[0] = f2bf(v.x); dst[1] = f2bf(v.y); dst[2] = f2bf(v.z); dst[3] = f2bf(v.w);
      }
      __syncthreads();
      const int halfl = lane & 15;
      const int ak = (lane >= 16) ? 8 : 0;
#pragma unroll
      for (int s = 0; s < 2; ++s) {
        const unsigned short* src = &As[(wm * 32 + s * 16 + halfl) * 40];
        fa[s].q[0] = *(const uint4*)(src + ak);
        fa[s].q[1] = *(const uint4*)(src + ak + 16);
      }
    } else {
      // A fragments straight from global (32B contiguous per lane, fully coalesced)
#pragma unroll
      for (int s = 0; s < 2; ++s) {
        const int mt = (m0 + wm * 32 + s * 16) >> 4;
        const unsigned short* src = Asw + (((size_t)mt * KT + kt) * 32 + lane) * 16;
        fa[s].q[0] = *(const uint4*)(src);
        fa[s].q[1] = *(const uint4*)(src + 8);
        __builtin_prefetch(src + 512, 0, 1);  // next k-tile of this m-tile
      }
    }
    // B fragments straight from global (32B contiguous per lane)
#pragma unroll
    for (int t = 0; t < 2; ++t) {
      const int nt = (n0 + wn * 32 + t * 16) >> 4;
      const unsigned short* src = Bsw + (((size_t)kt * NT4 + nt) * 32 + lane) * 16;
      fb[t].q[0] = *(const uint4*)(src);
      fb[t].q[1] = *(const uint4*)(src + 8);
    }
#pragma unroll
    for (int s = 0; s < 2; ++s)
#pragma unroll
      for (int t = 0; t < 2; ++t)
        acc[s][t] = __builtin_amdgcn_wmma_f32_16x16x32_bf16(
            false, fa[s].v, false, fb[t].v, (short)0, acc[s][t], false, false);
    if constexpr (AMODE == 0) __syncthreads();
  }

  // epilogue: C 16x16 f32 frag: lanes0-15 N=lane, reg r -> M=r; lanes16-31 M=8+r
  const int nloc = lane & 15;
  const int mofs = (lane >= 16) ? 8 : 0;
#pragma unroll
  for (int s = 0; s < 2; ++s) {
#pragma unroll
    for (int t = 0; t < 2; ++t) {
      const int gcol = n0 + wn * 32 + t * 16 + nloc;
      float bv = 0.f;
      if constexpr (HASBIAS) bv = bias[gcol];  // bias pre-padded to N, unconditional load
#pragma unroll
      for (int r = 0; r < 8; ++r) {
        const int grow = m0 + wm * 32 + s * 16 + mofs + r;
        float v = (acc[s][t][r] + bv) * scale;
        if constexpr (RELU) v = fmaxf(v, 0.f);
        if constexpr (OUTM == 1) Cf[(size_t)grow * ldc + gcol] = v;
        else                     Cb[a_swz(grow, gcol, N)] = f2bf(v);
      }
    }
  }
}

// ---------------- host ----------------
extern "C" void kernel_launch(void* const* d_in, const int* in_sizes, int n_in,
                              void* d_out, int out_size, void* d_ws, size_t ws_size,
                              hipStream_t stream) {
  const int* x[2]    = {(const int*)d_in[0], (const int*)d_in[5]};
  const int* ei[2]   = {(const int*)d_in[1], (const int*)d_in[6]};
  const int* ea[2]   = {(const int*)d_in[2], (const int*)d_in[7]};
  const int* bid[2]  = {(const int*)d_in[3], (const int*)d_in[8]};
  const int* dang[2] = {(const int*)d_in[4], (const int*)d_in[9]};
  const float* ae1 = (const float*)d_in[10];
  const float* ae2 = (const float*)d_in[11];
  const float* ee1 = (const float*)d_in[12];
  const float* ee2 = (const float*)d_in[13];
  const float* W1  = (const float*)d_in[14];
  const float* b1  = (const float*)d_in[15];
  const float* W2  = (const float*)d_in[16];
  const float* b2  = (const float*)d_in[17];
  const float* bng = (const float*)d_in[18];
  const float* bnb = (const float*)d_in[19];
  const float* pW1 = (const float*)d_in[20];
  const float* pb1 = (const float*)d_in[21];
  const float* pW2 = (const float*)d_in[22];
  const float* pb2 = (const float*)d_in[23];
  float* out = (float*)d_out;

  char* w = (char*)d_ws;
  auto take = [&](size_t bytes) -> void* {
    void* p = (void*)w;
    w += (bytes + 255) & ~(size_t)255;
    return p;
  };
  float*          hbuf   = (float*)take((size_t)NP * DP * 4);
  float*          agg    = (float*)take((size_t)NP * DP * 4);
  unsigned short* z1     = (unsigned short*)take((size_t)NP * HP * 2);  // A-swizzled
  unsigned short* W1b    = (unsigned short*)take((size_t)LL * DP * HP * 2);
  unsigned short* W2b    = (unsigned short*)take((size_t)LL * HP * DP * 2);
  unsigned short* pW1b   = (unsigned short*)take((size_t)4 * DP * DP * 2);
  unsigned short* pW2b   = (unsigned short*)take((size_t)4 * DP * DP * 2);
  float*          b1p    = (float*)take((size_t)LL * HP * 4);
  float*          b2p    = (float*)take((size_t)LL * DP * 4);
  float*          pb1p   = (float*)take((size_t)4 * DP * 4);
  float*          pb2p   = (float*)take((size_t)4 * DP * 4);
  float*          colsum = (float*)take((size_t)DP * 4);
  float*          colsq  = (float*)take((size_t)DP * 4);
  float*          pooled = (float*)take((size_t)NG * DP * 4);
  float*          counts = (float*)take((size_t)NG * 4);
  float*          hd     = (float*)take((size_t)NG * DP * 4);
  unsigned short* tbf    = (unsigned short*)take((size_t)NG * DP * 2);  // A-swizzled
  float*          o1     = (float*)take((size_t)NG * DP * 4);
  float*          o2     = (float*)take((size_t)NG * DP * 4);
  unsigned short* f0b    = (unsigned short*)take((size_t)NG * DP * 2);  // A-swizzled
  unsigned short* f1t    = (unsigned short*)take((size_t)DP * NG * 2);  // B-swizzled

  const int TB = 256;
  const size_t SHA = 128 * 40 * 2;  // dynamic LDS for AMODE==0 GEMMs

  // weight / bias prep (every call; deterministic)
  k_cvt_b<<<cdiv((long)LL * DP * HP, TB), TB, 0, stream>>>(W1, W1b, LL, 300, 600, DP, HP);
  k_cvt_b<<<cdiv((long)LL * HP * DP, TB), TB, 0, stream>>>(W2, W2b, LL, 600, 300, HP, DP);
  k_cvt_b<<<cdiv((long)4 * DP * DP, TB), TB, 0, stream>>>(pW1, pW1b, 4, 300, 300, DP, DP);
  k_cvt_b<<<cdiv((long)4 * DP * DP, TB), TB, 0, stream>>>(pW2, pW2b, 4, 300, 300, DP, DP);
  k_padbias<<<cdiv((long)LL * HP, TB), TB, 0, stream>>>(b1, b1p, LL, 600, HP);
  k_padbias<<<cdiv((long)LL * DP, TB), TB, 0, stream>>>(b2, b2p, LL, 300, DP);
  k_padbias<<<cdiv((long)4 * DP, TB), TB, 0, stream>>>(pb1, pb1p, 4, 300, DP);
  k_padbias<<<cdiv((long)4 * DP, TB), TB, 0, stream>>>(pb2, pb2p, 4, 300, DP);

  for (int bi = 0; bi < 2; ++bi) {
    k_zero<<<cdiv((long)NP * DP, TB), TB, 0, stream>>>(hbuf, NP * DP);
    k_zero<<<cdiv((long)NP * DP, TB), TB, 0, stream>>>(agg, NP * DP);
    k_init_h<<<cdiv((long)NN * DD, TB), TB, 0, stream>>>(x[bi], ae1, ae2, hbuf);

    for (int l = 0; l < LL; ++l) {
      k_agg_init<<<cdiv((long)NN * DD, TB), TB, 0, stream>>>(
          hbuf, ee1 + (size_t)(l * 6 + 4) * DD, ee2 + (size_t)(l * 3 + 0) * DD, agg);
      k_edge<<<cdiv((long)NE * 75, TB), TB, 0, stream>>>(
          ei[bi], ea[bi], hbuf, ee1 + (size_t)l * 6 * DD, ee2 + (size_t)l * 3 * DD, agg);
      // z1 = relu(agg @ W1 + b1)  -> bf16 A-swizzled [NP, HP]
      dim3 g1(HP / 64, NP / 128);
      k_gemm<0, true, true, 2><<<g1, 256, SHA, stream>>>(
          agg, W1b + (size_t)l * DP * HP, b1p + (size_t)l * HP, nullptr, z1,
          NP, HP, DP, DP, 0, 1.f);
      // z = z1 @ W2 + b2 -> agg (f32 row-major), LDS-free GEMM
      dim3 g2(DP / 64, NP / 128);
      k_gemm<1, false, true, 1><<<g2, 256, 0, stream>>>(
          z1, W2b + (size_t)l * HP * DP, b2p + (size_t)l * DP, agg, nullptr,
          NP, DP, HP, 0, DP, 1.f);
      // batchnorm over nodes -> hbuf (+relu except last layer)
      k_zero<<<2, TB, 0, stream>>>(colsum, DP);
      k_zero<<<2, TB, 0, stream>>>(colsq, DP);
      k_bn_stats<<<3072, 256, 0, stream>>>(agg, colsum, colsq);
      k_bn_apply<<<cdiv((long)NN * DD, TB), TB, 0, stream>>>(
          agg, colsum, colsq, bng + (size_t)l * DD, bnb + (size_t)l * DD, hbuf,
          (l < LL - 1) ? 1 : 0);
    }

    // branch: mean-pool + projector MLPs + dangling MLP + L2 normalize
    k_zero<<<cdiv((long)NG * DP, TB), TB, 0, stream>>>(pooled, NG * DP);
    k_zero<<<cdiv((long)NG, TB), TB, 0, stream>>>(counts, NG);
    k_counts<<<cdiv((long)NN, TB), TB, 0, stream>>>(bid[bi], counts);
    k_pool<<<cdiv((long)NN * 75, TB), TB, 0, stream>>>(bid[bi], hbuf, pooled);
    k_pooldiv<<<cdiv((long)NG * DD, TB), TB, 0, stream>>>(counts, pooled);
    k_gather<<<cdiv((long)NG * DP, TB), TB, 0, stream>>>(dang[bi], hbuf, hd);

    const int pi = bi, di = 2 + bi;
    dim3 gp(DP / 64, NG / 128);
    k_gemm<0, true, true, 2><<<gp, 256, SHA, stream>>>(
        pooled, pW1b + (size_t)pi * DP * DP, pb1p + (size_t)pi * DP, nullptr, tbf,
        NG, DP, DP, DP, 0, 1.f);
    k_gemm<1, false, true, 1><<<gp, 256, 0, stream>>>(
        tbf, pW2b + (size_t)pi * DP * DP, pb2p + (size_t)pi * DP, o1, nullptr,
        NG, DP, DP, 0, DP, 1.f);
    k_gemm<0, true, true, 2><<<gp, 256, SHA, stream>>>(
        hd, pW1b + (size_t)di * DP * DP, pb1p + (size_t)di * DP, nullptr, tbf,
        NG, DP, DP, DP, 0, 1.f);
    k_gemm<1, false, true, 1><<<gp, 256, 0, stream>>>(
        tbf, pW2b + (size_t)di * DP * DP, pb2p + (size_t)di * DP, o2, nullptr,
        NG, DP, DP, 0, DP, 1.f);
    k_norm<<<NG, 256, 0, stream>>>(o1, o2, f0b, f1t, bi);
  }

  // logits = (f0 @ f1^T) / TEMP -> d_out [4096, 4096] f32, fully LDS-free GEMM
  dim3 gl(NG / 64, NG / 128);
  k_gemm<1, false, false, 1><<<gl, 256, 0, stream>>>(
      f0b, f1t, nullptr, out, nullptr, NG, NG, DP, 0, NG, 25.f);
}
